// CGCNN_37855841747613
// MI455X (gfx1250) — matrix operand order repaired and checked by reading, not compiled, not run on hardware
//
#include <hip/hip_runtime.h>
#include <hip/hip_bf16.h>
#include <stdint.h>

typedef __bf16 bf16_t;
typedef __attribute__((ext_vector_type(16))) __bf16 v16bf;
typedef __attribute__((ext_vector_type(8)))  float  v8f;

#define NN      40000
#define NE      640000
#define NODEF   92
#define EDGEF   64
#define HD      128
#define LCONV   4
#define BN_EPS  1e-3f
#define MAXTILES (NN + NE / 16)   // bound on sum of ceil(deg/16) = 80000

static __device__ __forceinline__ v8f wmma_bf16(v16bf a, v16bf b, v8f c) {
  // D = A(16x32 bf16) x B(32x16 bf16) + C(16x16 f32)
  return __builtin_amdgcn_wmma_f32_16x16x32_bf16(false, a, false, b, (short)0, c, false, false);
}

// ---------------- utility kernels ----------------

__global__ void zero_kernel(float* __restrict__ p, long n) {
  long i = (long)blockIdx.x * blockDim.x + threadIdx.x;
  long stride = (long)gridDim.x * blockDim.x;
  for (; i < n; i += stride) p[i] = 0.f;
}

// WembT[j][k], j<128, k<96 (zero-pad k>=92), from W_emb[k][j]
__global__ void cvt_wemb(const float* __restrict__ W_emb, bf16_t* __restrict__ WembT) {
  int idx = blockIdx.x * blockDim.x + threadIdx.x;
  if (idx >= 128 * 96) return;
  int j = idx / 96, k = idx % 96;
  WembT[idx] = (k < NODEF) ? (bf16_t)W_emb[k * HD + j] : (bf16_t)0.f;
}

// WeBt[col(512)][k(64)]; col = l*128+j; from We[l][k][j]
__global__ void cvt_we(const float* __restrict__ We, bf16_t* __restrict__ WeBt) {
  int idx = blockIdx.x * blockDim.x + threadIdx.x;
  if (idx >= 512 * EDGEF) return;
  int col = idx / EDGEF, k = idx % EDGEF;
  int l = col >> 7, j = col & 127;
  WeBt[idx] = (bf16_t)We[((size_t)l * EDGEF + k) * HD + j];
}

// WnT[l][j][k] from Wn[l][k][j]
__global__ void cvt_wn(const float* __restrict__ Wn, bf16_t* __restrict__ WnT) {
  int idx = blockIdx.x * blockDim.x + threadIdx.x;
  if (idx >= LCONV * HD * HD) return;
  int l = idx >> 14, r = idx & 16383, j = r >> 7, k = r & 127;
  WnT[idx] = (bf16_t)Wn[((size_t)l * HD + k) * HD + j];
}

// ---------------- CSR build (sort edges by destination node) ----------------

__global__ void count_kernel(const int* __restrict__ edge_row, int* __restrict__ cnt) {
  int e = blockIdx.x * blockDim.x + threadIdx.x;
  if (e < NE) atomicAdd(&cnt[edge_row[e]], 1);
}

#define SCAN_B 1024
// exclusive prefix sums of cnt (edges) and ceil(cnt/16) (tiles), single block
__global__ __launch_bounds__(SCAN_B) void scan_kernel(
    const int* __restrict__ cnt, int* __restrict__ csr_start,
    int* __restrict__ tile_start, int* __restrict__ total) {
  __shared__ int s_e[SCAN_B];
  __shared__ int s_t[SCAN_B];
  __shared__ int carry_e, carry_t;
  int tid = threadIdx.x;
  if (tid == 0) { carry_e = 0; carry_t = 0; }
  __syncthreads();
  for (int base = 0; base < NN; base += SCAN_B) {
    int i = base + tid;
    int c = (i < NN) ? cnt[i] : 0;
    int t = (c + 15) >> 4;
    s_e[tid] = c; s_t[tid] = t;
    __syncthreads();
    for (int off = 1; off < SCAN_B; off <<= 1) {
      int ae = (tid >= off) ? s_e[tid - off] : 0;
      int at = (tid >= off) ? s_t[tid - off] : 0;
      __syncthreads();
      s_e[tid] += ae; s_t[tid] += at;
      __syncthreads();
    }
    if (i < NN) {
      csr_start[i]  = carry_e + s_e[tid] - c;   // exclusive
      tile_start[i] = carry_t + s_t[tid] - t;
    }
    __syncthreads();
    if (tid == SCAN_B - 1) { carry_e += s_e[tid]; carry_t += s_t[tid]; }
    __syncthreads();
  }
  if (tid == 0) { csr_start[NN] = carry_e; tile_start[NN] = carry_t; total[0] = carry_t; }
}

// one entry per 16-edge tile: meta = (node<<5)|valid, ebase = first slot in perm
__global__ void fill_kernel(const int* __restrict__ cnt, const int* __restrict__ csr_start,
                            const int* __restrict__ tile_start,
                            int* __restrict__ tile_meta, int* __restrict__ tile_ebase) {
  int n = blockIdx.x * blockDim.x + threadIdx.x;
  if (n >= NN) return;
  int deg = cnt[n], ts = tile_start[n], cs = csr_start[n];
  int ntiles = (deg + 15) >> 4;
  for (int c = 0; c < ntiles; ++c) {
    int valid = deg - c * 16; if (valid > 16) valid = 16;
    tile_meta[ts + c]  = (n << 5) | valid;
    tile_ebase[ts + c] = cs + c * 16;
  }
}

__global__ void scatter_kernel(const int* __restrict__ edge_row,
                               const int* __restrict__ csr_start,
                               int* __restrict__ fill, int* __restrict__ perm) {
  int e = blockIdx.x * blockDim.x + threadIdx.x;
  if (e >= NE) return;
  int r = edge_row[e];
  int pos = atomicAdd(&fill[r], 1);
  perm[csr_start[r] + pos] = e;
}

// ---------------- h0 = x @ W_emb + b_emb ----------------

__global__ __launch_bounds__(256) void embed_kernel(
    const float* __restrict__ x, const bf16_t* __restrict__ WembT,
    const float* __restrict__ b_emb, float* __restrict__ h) {
  int lane = threadIdx.x & 31;
  int wave = threadIdx.x >> 5;
  int tile = blockIdx.x * 8 + wave;        // uniform per wave
  if (tile >= NN / 16) return;
  int half = lane >> 4, mr = lane & 15;
  int n0 = tile * 16;
  const float* arow = x + (size_t)(n0 + mr) * NODEF;

  v16bf A[3];
  #pragma unroll
  for (int ks = 0; ks < 3; ++ks) {
    v16bf a;
    #pragma unroll
    for (int e = 0; e < 16; ++e) {
      int k = e + half * 16 + ks * 32;
      float vv = (k < NODEF) ? arow[k] : 0.f;
      a[e] = (bf16_t)vv;
    }
    A[ks] = a;
  }
  #pragma unroll
  for (int ct = 0; ct < 8; ++ct) {
    int j = ct * 16 + mr;
    v8f acc = {};
    #pragma unroll
    for (int ks = 0; ks < 3; ++ks) {
      v16bf b = *(const v16bf*)(WembT + (size_t)j * 96 + half * 16 + ks * 32);
      acc = wmma_bf16(A[ks], b, acc);
    }
    float bias = b_emb[j];
    #pragma unroll
    for (int i = 0; i < 8; ++i) {
      int m = i + half * 8;
      h[(size_t)(n0 + m) * HD + j] = acc[i] + bias;
    }
  }
}

// ------- S[l][n][:] = seg-sum relu(edge_attr@We[l]+be[l]); per-node 16-edge tiles -------

#define TPB_EDGE 8   // tiles per block

__global__ __launch_bounds__(256) void edge_kernel(
    const float* __restrict__ edge_attr, const int* __restrict__ perm,
    const int* __restrict__ tile_meta, const int* __restrict__ tile_ebase,
    const int* __restrict__ total_tiles,
    const bf16_t* __restrict__ WeBt, const float* __restrict__ be,
    float* __restrict__ S) {
  __shared__ float atile[16 * 68];         // padded stride: conflict-free
  int lane = threadIdx.x & 31;
  int wave = threadIdx.x >> 5;             // 8 waves x 64 cols = 512 cols
  int half = lane >> 4, mr = lane & 15;
  int colbase = wave * 64;
  int T = total_tiles[0];

  v16bf Bf[4][2];
  float bias[4];
  #pragma unroll
  for (int ct = 0; ct < 4; ++ct) {
    int n = colbase + ct * 16 + mr;
    bias[ct] = be[n];                      // be flat [4*128] == stacked cols
    #pragma unroll
    for (int ks = 0; ks < 2; ++ks)
      Bf[ct][ks] = *(const v16bf*)(WeBt + (size_t)n * EDGEF + half * 16 + ks * 32);
  }

  for (int t = 0; t < TPB_EDGE; ++t) {
    int tile = blockIdx.x * TPB_EDGE + t;  // uniform across block
    if (tile >= T) break;
    int meta = tile_meta[tile];
    int node = meta >> 5, valid = meta & 31;
    int ebase = tile_ebase[tile];

    // stage gathered 16x64 A tile (rows = this node's edges) into LDS
    #pragma unroll
    for (int i2 = 0; i2 < 4; ++i2) {
      int f = threadIdx.x + 256 * i2;
      int r = f >> 6, c = f & 63;
      float v = 0.f;
      if (r < valid) {
        int e = perm[ebase + r];
        v = edge_attr[(size_t)e * EDGEF + c];
      }
      atile[r * 68 + c] = v;
    }
    __syncthreads();

    v16bf A[2];
    #pragma unroll
    for (int ks = 0; ks < 2; ++ks) {
      v16bf a;
      #pragma unroll
      for (int e = 0; e < 16; ++e) {
        int k = e + half * 16 + ks * 32;
        a[e] = (bf16_t)atile[mr * 68 + k];
      }
      A[ks] = a;
    }

    v8f acc[4];
    #pragma unroll
    for (int ct = 0; ct < 4; ++ct) {
      v8f c = {};
      c = wmma_bf16(A[0], Bf[ct][0], c);
      c = wmma_bf16(A[1], Bf[ct][1], c);
      acc[ct] = c;
    }

    // epilogue: bias+relu per element, mask padded rows, column-sum in registers
    #pragma unroll
    for (int ct = 0; ct < 4; ++ct) {
      int n = colbase + ct * 16 + mr;
      int layer = n >> 7, j = n & 127;
      float s = 0.f;
      #pragma unroll
      for (int i = 0; i < 8; ++i) {
        int m = i + half * 8;
        float d = fmaxf(acc[ct][i] + bias[ct], 0.f);
        s += (m < valid) ? d : 0.f;
      }
      s += __shfl_xor(s, 16, 32);          // combine the two 8-row half-wave groups
      if (lane < 16)                       // one atomic per column per tile
        unsafeAtomicAdd(&S[((size_t)layer * NN + node) * HD + j], s);
    }
    __syncthreads();
  }
}

// ------- h += BN(relu((h .* S_l) @ Wn_l + bn_l)) -------

__global__ __launch_bounds__(256) void conv_kernel(
    float* __restrict__ h, const float* __restrict__ S_l,
    const bf16_t* __restrict__ WnT_l, const float* __restrict__ bn_l,
    const float* __restrict__ g_l, const float* __restrict__ beta_l,
    const float* __restrict__ mu_l, const float* __restrict__ var_l) {
  int lane = threadIdx.x & 31;
  int wave = threadIdx.x >> 5;
  int tile = blockIdx.x * 8 + wave;
  if (tile >= NN / 16) return;
  int half = lane >> 4, mr = lane & 15;
  int n0 = tile * 16;
  const float* hrow = h + (size_t)(n0 + mr) * HD;
  const float* srow = S_l + (size_t)(n0 + mr) * HD;

  v16bf A[4];
  #pragma unroll
  for (int ks = 0; ks < 4; ++ks) {
    v16bf a;
    #pragma unroll
    for (int e = 0; e < 16; ++e) {
      int k = e + half * 16 + ks * 32;
      a[e] = (bf16_t)(hrow[k] * srow[k]);
    }
    A[ks] = a;
  }
  #pragma unroll
  for (int ct = 0; ct < 8; ++ct) {
    int j = ct * 16 + mr;
    v8f acc = {};
    #pragma unroll
    for (int ks = 0; ks < 4; ++ks) {
      v16bf b = *(const v16bf*)(WnT_l + (size_t)j * HD + half * 16 + ks * 32);
      acc = wmma_bf16(A[ks], b, acc);
    }
    float bias = bn_l[j];
    float scale = g_l[j] * rsqrtf(var_l[j] + BN_EPS);
    float mu = mu_l[j], beta = beta_l[j];
    #pragma unroll
    for (int i = 0; i < 8; ++i) {
      int m = i + half * 8;
      float u = fmaxf(acc[i] + bias, 0.f);
      u = (u - mu) * scale + beta;
      h[(size_t)(n0 + m) * HD + j] += u;   // residual
    }
  }
}

// ---------------- pool + dense head ----------------

__global__ void pool_kernel(const float* __restrict__ h, float* __restrict__ vsum) {
  int j = threadIdx.x & 127;
  int chunk = blockIdx.x * 2 + (threadIdx.x >> 7);
  float s = 0.f;
  for (int n = chunk; n < NN; n += gridDim.x * 2)
    s += h[(size_t)n * HD + j];
  unsafeAtomicAdd(&vsum[j], s);
}

__global__ void head_kernel(const float* __restrict__ vsum,
                            const float* __restrict__ Wd, const float* __restrict__ bd,
                            const float* __restrict__ g_d, const float* __restrict__ beta_d,
                            const float* __restrict__ mu_d, const float* __restrict__ var_d,
                            const float* __restrict__ Wf, const float* __restrict__ bfv,
                            float* __restrict__ out) {
  __shared__ float v[HD];
  int j = threadIdx.x;
  v[j] = vsum[j] * (1.f / (float)NN);
  __syncthreads();
  for (int d = 0; d < 2; ++d) {
    float t = bd[d * HD + j];
    for (int k = 0; k < HD; ++k) t += v[k] * Wd[((size_t)d * HD + k) * HD + j];
    t = fmaxf(t, 0.f);
    t = (t - mu_d[d * HD + j]) * (g_d[d * HD + j] * rsqrtf(var_d[d * HD + j] + BN_EPS))
        + beta_d[d * HD + j];
    __syncthreads();
    v[j] = t;
    __syncthreads();
  }
  if (j < 3) {
    float t = bfv[j];
    for (int k = 0; k < HD; ++k) t += v[k] * Wf[k * 3 + j];
    out[j] = t;
  }
}

// ---------------- launch ----------------

extern "C" void kernel_launch(void* const* d_in, const int* in_sizes, int n_in,
                              void* d_out, int out_size, void* d_ws, size_t ws_size,
                              hipStream_t stream) {
  (void)in_sizes; (void)n_in; (void)out_size; (void)ws_size;
  const float* x         = (const float*)d_in[0];
  const int*   edge_idx  = (const int*)d_in[1];    // row = first NE entries
  const float* edge_attr = (const float*)d_in[2];
  const float* W_emb     = (const float*)d_in[3];
  const float* b_emb     = (const float*)d_in[4];
  const float* We        = (const float*)d_in[5];
  const float* be        = (const float*)d_in[6];
  const float* Wn        = (const float*)d_in[7];
  const float* bn        = (const float*)d_in[8];
  const float* g_c       = (const float*)d_in[9];
  const float* beta_c    = (const float*)d_in[10];
  const float* mu_c      = (const float*)d_in[11];
  const float* var_c     = (const float*)d_in[12];
  const float* Wd        = (const float*)d_in[13];
  const float* bd        = (const float*)d_in[14];
  const float* g_d       = (const float*)d_in[15];
  const float* beta_d    = (const float*)d_in[16];
  const float* mu_d      = (const float*)d_in[17];
  const float* var_d     = (const float*)d_in[18];
  const float* Wf        = (const float*)d_in[19];
  const float* bf_       = (const float*)d_in[20];
  float* out = (float*)d_out;

  char* ws = (char*)d_ws;
  float*  h          = (float*)(ws);                 // NN*HD f32        20,480,000 B
  float*  S          = (float*)(ws + 20480000);      // 4*NN*HD f32      81,920,000 B
  float*  vsum       = (float*)(ws + 102400000);     // 128 f32                512 B
  int*    cnt        = (int*)  (ws + 102400512);     // NN                 160,000 B
  int*    fill       = (int*)  (ws + 102560512);     // NN                 160,000 B
  int*    csr_start  = (int*)  (ws + 102720512);     // NN+1 (pad 160,256)
  int*    tile_start = (int*)  (ws + 102880768);     // NN+1 (pad 160,256)
  int*    total      = (int*)  (ws + 103041024);     // 1 (pad 256)
  int*    perm       = (int*)  (ws + 103041280);     // NE               2,560,000 B
  int*    tile_meta  = (int*)  (ws + 105601280);     // MAXTILES           320,000 B
  int*    tile_ebase = (int*)  (ws + 105921280);     // MAXTILES           320,000 B
  bf16_t* WembT      = (bf16_t*)(ws + 106241280);    // 128*96 bf16
  bf16_t* WeBt       = (bf16_t*)(ws + 106265856);    // 512*64 bf16
  bf16_t* WnT        = (bf16_t*)(ws + 106331392);    // 4*128*128 bf16 -> end 106,462,464

  // S, vsum, cnt, fill are contiguous -> one zeroing pass (82,240,512 B)
  zero_kernel<<<512, 256, 0, stream>>>(S, 20560128L);
  cvt_wemb<<<(128 * 96 + 255) / 256, 256, 0, stream>>>(W_emb, WembT);
  cvt_we<<<(512 * EDGEF + 255) / 256, 256, 0, stream>>>(We, WeBt);
  cvt_wn<<<(LCONV * HD * HD + 255) / 256, 256, 0, stream>>>(Wn, WnT);

  // CSR build: count -> scan -> tile fill -> scatter
  count_kernel<<<(NE + 255) / 256, 256, 0, stream>>>(edge_idx, cnt);
  scan_kernel<<<1, SCAN_B, 0, stream>>>(cnt, csr_start, tile_start, total);
  fill_kernel<<<(NN + 255) / 256, 256, 0, stream>>>(cnt, csr_start, tile_start,
                                                    tile_meta, tile_ebase);
  scatter_kernel<<<(NE + 255) / 256, 256, 0, stream>>>(edge_idx, csr_start, fill, perm);

  embed_kernel<<<(NN / 16 + 7) / 8, 256, 0, stream>>>(x, WembT, b_emb, h);
  edge_kernel<<<(MAXTILES + TPB_EDGE - 1) / TPB_EDGE, 256, 0, stream>>>(
      edge_attr, perm, tile_meta, tile_ebase, total, WeBt, be, S);

  for (int l = 0; l < LCONV; ++l) {
    conv_kernel<<<(NN / 16 + 7) / 8, 256, 0, stream>>>(
        h, S + (size_t)l * NN * HD, WnT + (size_t)l * HD * HD,
        bn + l * HD, g_c + l * HD, beta_c + l * HD, mu_c + l * HD, var_c + l * HD);
  }
  pool_kernel<<<200, 256, 0, stream>>>(h, vsum);
  head_kernel<<<1, HD, 0, stream>>>(vsum, Wd, bd, g_d, beta_d, mu_d, var_d, Wf, bf_, out);
}
